// SelfAttention_1090921693491
// MI455X (gfx1250) — matrix operand order repaired and checked by reading, not compiled, not run on hardware
//
#include <hip/hip_runtime.h>

typedef __attribute__((ext_vector_type(16))) _Float16 v16h;
typedef __attribute__((ext_vector_type(8)))  _Float16 v8h;
typedef __attribute__((ext_vector_type(8)))  float    v8f;

#define B_SZ 2
#define C_SZ 64
#define C8_SZ 8
#define N_SEQ 9216   // 96*96
#define LOG2E 1.4426950408889634f

__device__ __forceinline__ v8h ldh8(const _Float16* p) {
    return *(const v8h*)p;
}

// ---------------------------------------------------------------------------
// Kernel 1: spectral norm sigma via power iteration + zero-page init.
// blockIdx.x = 0 (Wq 8x64), 1 (Wk 8x64), 2 (Wv 64x64). 64 threads.
// ---------------------------------------------------------------------------
__global__ __launch_bounds__(64) void sigma_kernel(const float* __restrict__ Wq,
                                                   const float* __restrict__ Wk,
                                                   const float* __restrict__ Wv,
                                                   float* __restrict__ sig,
                                                   float* __restrict__ zpage) {
    __shared__ float v[64];
    __shared__ float u[64];
    __shared__ float red[64];
    const int m = blockIdx.x;
    const float* W = (m == 0) ? Wq : (m == 1) ? Wk : Wv;
    const int R = (m == 2) ? 64 : 8;
    const int t = threadIdx.x;

    if (m == 0 && t < 16) zpage[t] = 0.0f;   // 64B zero page for attn A-operand

    v[t] = 1.0f;
    __syncthreads();

    for (int it = 0; it < 32; ++it) {
        if (t < R) {
            float acc = 0.f;
            for (int c = 0; c < 64; ++c) acc += W[t * 64 + c] * v[c];
            u[t] = acc;
        }
        __syncthreads();
        float acc = 0.f;
        for (int o = 0; o < R; ++o) acc += W[o * 64 + t] * u[o];
        red[t] = acc * acc;
        __syncthreads();
        for (int s = 32; s > 0; s >>= 1) {
            if (t < s) red[t] += red[t + s];
            __syncthreads();
        }
        float nrm = sqrtf(red[0]) + 1e-20f;
        __syncthreads();
        v[t] = acc / nrm;
        __syncthreads();
    }
    // sigma = ||W v||  (||v|| == 1)
    if (t < R) {
        float acc = 0.f;
        for (int c = 0; c < 64; ++c) acc += W[t * 64 + c] * v[c];
        red[t] = acc * acc;
    } else {
        red[t] = 0.f;
    }
    __syncthreads();
    for (int s = 32; s > 0; s >>= 1) {
        if (t < s) red[t] += red[t + s];
        __syncthreads();
    }
    if (t == 0) sig[m] = sqrtf(red[0]) + 1e-20f;
}

// ---------------------------------------------------------------------------
// Kernel 2: fused f/g/h projections into f16 workspace.
//   fW[b][n][8] (pre-scaled by log2e so softmax uses raw v_exp_f32),
//   gW[b][n][8], hW[b][c][n].
// One block = 128 n's of one batch; x tile + scaled weights staged in LDS.
// ---------------------------------------------------------------------------
__global__ __launch_bounds__(128) void proj_kernel(const float* __restrict__ x,
                                                   const float* __restrict__ Wq,
                                                   const float* __restrict__ bq,
                                                   const float* __restrict__ Wk,
                                                   const float* __restrict__ bk,
                                                   const float* __restrict__ Wv,
                                                   const float* __restrict__ bv,
                                                   const float* __restrict__ sig,
                                                   _Float16* __restrict__ fW,
                                                   _Float16* __restrict__ gW,
                                                   _Float16* __restrict__ hW) {
    __shared__ float x_s[C_SZ * 128];       // 32 KB
    __shared__ float wv_s[C_SZ * C_SZ];     // 16 KB
    __shared__ float wq_s[C8_SZ * C_SZ];    //  2 KB
    __shared__ float wk_s[C8_SZ * C_SZ];    //  2 KB

    const int tid = threadIdx.x;
    const int b = blockIdx.y;
    const int n0 = blockIdx.x * 128;
    const float inv_q = 1.0f / sig[0];
    const float inv_k = 1.0f / sig[1];
    const float inv_v = 1.0f / sig[2];

    // stage x tile: x_s[c][nn]
    for (int k = tid; k < C_SZ * 128; k += 128) {
        int c = k >> 7, nn = k & 127;
        x_s[k] = x[((size_t)b * C_SZ + c) * N_SEQ + n0 + nn];
    }
    // stage spectrally-normalized weights
    for (int k = tid; k < C_SZ * C_SZ; k += 128) wv_s[k] = Wv[k] * inv_v;
    for (int k = tid; k < C8_SZ * C_SZ; k += 128) {
        wq_s[k] = Wq[k] * inv_q;
        wk_s[k] = Wk[k] * inv_k;
    }
    __syncthreads();

    const int n = n0 + tid;

    // f and g
    float fa[C8_SZ], ga[C8_SZ];
#pragma unroll
    for (int o = 0; o < C8_SZ; ++o) { fa[o] = 0.f; ga[o] = 0.f; }
    for (int c = 0; c < C_SZ; ++c) {
        float xv = x_s[c * 128 + tid];
#pragma unroll
        for (int o = 0; o < C8_SZ; ++o) {
            fa[o] = fmaf(wq_s[o * 64 + c], xv, fa[o]);
            ga[o] = fmaf(wk_s[o * 64 + c], xv, ga[o]);
        }
    }
#pragma unroll
    for (int o = 0; o < C8_SZ; ++o) {
        // fold log2(e) into f so attn softmax can use v_exp_f32 directly
        fW[((size_t)b * N_SEQ + n) * C8_SZ + o] = (_Float16)((fa[o] + bq[o]) * LOG2E);
        gW[((size_t)b * N_SEQ + n) * C8_SZ + o] = (_Float16)(ga[o] + bk[o]);
    }

    // h
    for (int co = 0; co < C_SZ; ++co) {
        float acc = bv[co];
#pragma unroll 8
        for (int c = 0; c < C_SZ; ++c)
            acc = fmaf(wv_s[co * 64 + c], x_s[c * 128 + tid], acc);
        hW[((size_t)b * C_SZ + co) * N_SEQ + n] = (_Float16)acc;
    }
}

// ---------------------------------------------------------------------------
// Kernel 3: flash attention (log2-domain softmax).
//   query j <- g[:,j] (d=8), key i <- f[:,i] (pre-scaled log2e), value h[:,i].
//   4 waves/WG; each wave owns 16 query columns.
//   Per 32-key step: 2 QK^T WMMAs (K padded 8->32) + 4 PV WMMAs (c tiles).
// ---------------------------------------------------------------------------
__global__ __launch_bounds__(128) void attn_kernel(const float* __restrict__ x,
                                                   const float* __restrict__ gamma,
                                                   const _Float16* __restrict__ fW,
                                                   const _Float16* __restrict__ gW,
                                                   const _Float16* __restrict__ hW,
                                                   const _Float16* __restrict__ zpage,
                                                   float* __restrict__ out) {
    const int b = blockIdx.y;
    const int lane = threadIdx.x & 31;
    const int wave = threadIdx.x >> 5;
    const int j0 = blockIdx.x * 64 + wave * 16;
    const int hl = lane & 15;
    const bool hi = lane >= 16;

    // B operand (g) for this wave's 16 query columns; K = c in [0,8), rest 0.
    v8h gv = ldh8(gW + ((size_t)b * N_SEQ + j0 + hl) * C8_SZ);
    v16h bg = {};
#pragma unroll
    for (int e = 0; e < 8; ++e) bg[e] = hi ? (_Float16)0.0f : gv[e];

    const _Float16* fbase = fW + (size_t)b * N_SEQ * C8_SZ;
    const _Float16* hbase = hW + (size_t)b * C_SZ * N_SEQ;

    v8f O[4];
#pragma unroll
    for (int t = 0; t < 4; ++t) O[t] = (v8f){};
    float m_run = -3.0e38f;
    float l_run = 0.0f;

    for (int i0 = 0; i0 < N_SEQ; i0 += 32) {
        // prefetch next key tile of f (CDNA5 global_prefetch path)
        if (i0 + 32 < N_SEQ)
            __builtin_prefetch(fbase + (size_t)(i0 + 32 + hl) * C8_SZ, 0, 1);

        // A operands (f rows -> key tile). Lanes >= 16 read a zero page so no
        // per-element masking is needed; elements 8..15 are constant zeros.
        const _Float16* pf0 = hi ? zpage : fbase + (size_t)(i0 + hl) * C8_SZ;
        const _Float16* pf1 = hi ? zpage : fbase + (size_t)(i0 + 16 + hl) * C8_SZ;
        v8h f0 = ldh8(pf0);
        v8h f1 = ldh8(pf1);

        // issue h (value) loads early: independent of the score WMMAs, and
        // in-order load returns mean the f-wait does not wait on these.
        v8h hv0[4], hv1[4];
#pragma unroll
        for (int ct = 0; ct < 4; ++ct) {
            const _Float16* hp = hbase + (size_t)(ct * 16 + hl) * N_SEQ + i0;
            hv0[ct] = ldh8(hp + (hi ? 16 : 0));
            hv1[ct] = ldh8(hp + (hi ? 24 : 8));
        }

        v16h a0 = {}, a1 = {};
#pragma unroll
        for (int e = 0; e < 8; ++e) { a0[e] = f0[e]; a1[e] = f1[e]; }

        v8f zc = {};
        v8f S0 = __builtin_amdgcn_wmma_f32_16x16x32_f16(false, a0, false, bg,
                                                        (short)0, zc, false, false);
        v8f S1 = __builtin_amdgcn_wmma_f32_16x16x32_f16(false, a1, false, bg,
                                                        (short)0, zc, false, false);

        // online softmax over i (log2 domain; column j = lane pair {hl, hl+16})
        float tmax = -3.0e38f;
#pragma unroll
        for (int r = 0; r < 8; ++r)
            tmax = fmaxf(tmax, fmaxf(S0[r], S1[r]));
        tmax = fmaxf(tmax, __shfl_xor(tmax, 16, 32));
        float m_new = fmaxf(m_run, tmax);
        float scale = __builtin_amdgcn_exp2f(m_run - m_new);

        float p0[8], p1[8];
        float rowsum = 0.f;
#pragma unroll
        for (int r = 0; r < 8; ++r) {
            p0[r] = __builtin_amdgcn_exp2f(S0[r] - m_new);
            p1[r] = __builtin_amdgcn_exp2f(S1[r] - m_new);
            rowsum += p0[r] + p1[r];
        }
        rowsum += __shfl_xor(rowsum, 16, 32);
        l_run = l_run * scale + rowsum;
        m_run = m_new;

        // rescale accumulator
#pragma unroll
        for (int t = 0; t < 4; ++t)
#pragma unroll
            for (int r = 0; r < 8; ++r) O[t][r] *= scale;

        // Rearrange beta from D-tile layout to f16 B-operand layout (K=i).
        // Pack (p0,p1) as f16x2 (one v_cvt_pk_rtz) so ONE dword shuffle serves
        // both exchanges:
        //   lane<16 : K=r -> own p0[r]      ; K=8+r  -> partner p0[r] (lo half)
        //   lane>=16: K=16+r -> partner p1[r] (hi half) ; K=24+r -> own p1[r]
        v16h bp;
#pragma unroll
        for (int r = 0; r < 8; ++r) {
            int pki = __builtin_bit_cast(int, __builtin_amdgcn_cvt_pkrtz(p0[r], p1[r]));
            int oth = __shfl_xor(pki, 16, 32);
            unsigned short own_lo = (unsigned short)(pki & 0xffff);
            unsigned short own_hi = (unsigned short)(((unsigned)pki) >> 16);
            unsigned short oth_lo = (unsigned short)(oth & 0xffff);
            unsigned short oth_hi = (unsigned short)(((unsigned)oth) >> 16);
            bp[r]     = __builtin_bit_cast(_Float16, hi ? oth_hi : own_lo);
            bp[8 + r] = __builtin_bit_cast(_Float16, hi ? own_hi : oth_lo);
        }

        // PV: O[c,j] += h[c, i0..i0+32] * beta ; 4 c-tiles of 16
#pragma unroll
        for (int ct = 0; ct < 4; ++ct) {
            v16h ah;
#pragma unroll
            for (int e = 0; e < 8; ++e) { ah[e] = hv0[ct][e]; ah[8 + e] = hv1[ct][e]; }
            O[ct] = __builtin_amdgcn_wmma_f32_16x16x32_f16(false, ah, false, bp,
                                                           (short)0, O[ct],
                                                           false, false);
        }
    }

    // epilogue: out = gamma * O / l + x (residual)
    const float inv_l = 1.0f / l_run;
    const float gm = gamma[0];
#pragma unroll
    for (int ct = 0; ct < 4; ++ct) {
#pragma unroll
        for (int r = 0; r < 8; ++r) {
            int c = ct * 16 + (hi ? 8 : 0) + r;
            size_t idx = ((size_t)b * C_SZ + c) * N_SEQ + j0 + hl;
            out[idx] = gm * O[ct][r] * inv_l + x[idx];
        }
    }
}

// ---------------------------------------------------------------------------
extern "C" void kernel_launch(void* const* d_in, const int* in_sizes, int n_in,
                              void* d_out, int out_size, void* d_ws, size_t ws_size,
                              hipStream_t stream) {
    const float* x     = (const float*)d_in[0];
    const float* Wq    = (const float*)d_in[1];
    const float* bq    = (const float*)d_in[2];
    const float* Wk    = (const float*)d_in[3];
    const float* bk    = (const float*)d_in[4];
    const float* Wv    = (const float*)d_in[5];
    const float* bv    = (const float*)d_in[6];
    const float* gamma = (const float*)d_in[7];
    float* out = (float*)d_out;

    char* ws = (char*)d_ws;
    float* sig = (float*)ws;                                  // 3 floats
    float* zpage = (float*)(ws + 128);                        // 64B zeros
    _Float16* fW = (_Float16*)(ws + 256);                     // [B][N][8]
    _Float16* gW = fW + (size_t)B_SZ * N_SEQ * C8_SZ;         // [B][N][8]
    _Float16* hW = gW + (size_t)B_SZ * N_SEQ * C8_SZ;         // [B][C][N]
    // total workspace used: 256 + 2*294912 + 2359296 bytes ~= 2.95 MB

    sigma_kernel<<<dim3(3), dim3(64), 0, stream>>>(Wq, Wk, Wv, sig, zpage);
    proj_kernel<<<dim3(N_SEQ / 128, B_SZ), dim3(128), 0, stream>>>(
        x, Wq, bq, Wk, bk, Wv, bv, sig, fW, gW, hW);
    attn_kernel<<<dim3(N_SEQ / 64, B_SZ), dim3(128), 0, stream>>>(
        x, gamma, fW, gW, hW, (const _Float16*)zpage, out);
}
